// SegmentEncoder_51230369906918
// MI455X (gfx1250) — matrix operand order repaired and checked by reading, not compiled, not run on hardware
//
#include <hip/hip_runtime.h>
#include <stdint.h>

// MI455X / gfx1250: wave32, WMMA 16x16x32 f16 with f32 accumulate.

typedef _Float16 v16h __attribute__((ext_vector_type(16)));
typedef __fp16   v2fp __attribute__((ext_vector_type(2)));
typedef float    v8f  __attribute__((ext_vector_type(8)));

#define N_SEQ   131072
#define L_T     32
#define E_DIM   16
#define H_DIM   32
#define TILE_SEQ 16
#define WAVES_PER_BLK 4

// single v_cvt_pk_rtz_f16_f32
static __device__ __forceinline__ uint32_t pk2(float a, float b) {
  v2fp p = __builtin_amdgcn_cvt_pkrtz(a, b);
  return __builtin_bit_cast(uint32_t, p);
}

// v_rcp_f32: no IEEE-div slow path, no exec-mask branches
static __device__ __forceinline__ float fast_rcp(float x) {
  return __builtin_amdgcn_rcpf(x);
}
static __device__ __forceinline__ float sigm(float x) {
  return fast_rcp(1.0f + __expf(-x));
}
static __device__ __forceinline__ float tanh_f(float x) {
  const float e = __expf(-2.0f * x);
  return (1.0f - e) * fast_rcp(1.0f + e);
}

union FragAB { v16h h; uint32_t u[8]; };

static __device__ __forceinline__ v8f ldc8(const float* p) {
  v8f z;
#pragma unroll
  for (int i = 0; i < 8; ++i) z[i] = p[i];
  return z;
}

static __device__ __forceinline__ v8f wmma_hh(v16h a, v16h b, v8f c) {
  // D = A(16x32 f16) x B(32x16 f16) + C(16x16 f32)
  return __builtin_amdgcn_wmma_f32_16x16x32_f16(
      /*neg_a=*/false, a, /*neg_b=*/false, b,
      /*c_mod=*/(short)0, c, /*reuse_a=*/false, /*reuse_b=*/false);
}

// One direction of the masked LSTM. Each wave owns 16 sequences.
// Gate layout after WMMA (D layout): lane%16 = sequence-in-tile (N),
// VGPR r = gate unit j = 16*T + r + 8*(lane/16) (M).  State h/c kept as
// hA[r] = h[u = r+8*hi][seq], hB[r] = h[u = 16+r+8*hi][seq].
__global__ void __launch_bounds__(WAVES_PER_BLK * 32)
lstm_dir_kernel(const int* __restrict__ tokens, const int* __restrict__ lengths,
                const float* __restrict__ emb,  const float* __restrict__ W_ih,
                const float* __restrict__ W_hh, const float* __restrict__ bias,
                float* __restrict__ out, int dir)
{
  // Precomputed x-projection + bias for the 3 vocab entries: gxs[v][j]
  __shared__ float gxs[3 * 4 * H_DIM];   // 3 x 128 floats

  const int tid = threadIdx.x;
  for (int idx = tid; idx < 3 * 4 * H_DIM; idx += WAVES_PER_BLK * 32) {
    const int v = idx >> 7, j = idx & 127;
    float s = bias[j];
#pragma unroll
    for (int k = 0; k < E_DIM; ++k) s += emb[v * E_DIM + k] * W_ih[j * E_DIM + k];
    gxs[idx] = s;
  }
  __syncthreads();

  const int  lane = tid & 31;
  const int  m    = lane & 15;     // sequence-in-tile / row-in-tile
  const int  hi   = lane >> 4;     // which half-wave
  const long tile = (long)blockIdx.x * WAVES_PER_BLK + (tid >> 5);
  const int  seq  = (int)(tile * TILE_SEQ) + m;

  // ---- W_hh A-fragments (f16 A 16x32 layout): lane m holds row j=16T+m;
//      VGPR v<4: K = 16*hi + {2v,2v+1};  v>=4: K = 16*hi + 8 + {2(v-4),...}
  FragAB a_hh[8];
#pragma unroll
  for (int T = 0; T < 8; ++T) {
    const float* w = W_hh + (T * 16 + m) * H_DIM;
#pragma unroll
    for (int v = 0; v < 4; ++v) {
      const int u  = 16 * hi + 2 * v;
      const int u2 = 16 * hi + 8 + 2 * v;
      a_hh[T].u[v]     = pk2(w[u],  w[u + 1]);
      a_hh[T].u[v + 4] = pk2(w[u2], w[u2 + 1]);
    }
  }

  // ---- pack this lane's 32 tokens (each in {0,1,2}) into 64 bits
  unsigned long long tp = 0ull;
  {
    const int4* trow = (const int4*)(tokens + (long)seq * L_T);
#pragma unroll
    for (int q = 0; q < 8; ++q) {
      const int4 w4 = trow[q];
      tp |= (unsigned long long)(w4.x & 3) << (2 * (4 * q + 0));
      tp |= (unsigned long long)(w4.y & 3) << (2 * (4 * q + 1));
      tp |= (unsigned long long)(w4.z & 3) << (2 * (4 * q + 2));
      tp |= (unsigned long long)(w4.w & 3) << (2 * (4 * q + 3));
    }
  }
  const int len = lengths[seq];

  float hA[8], hB[8], cA[8], cB[8];
#pragma unroll
  for (int r = 0; r < 8; ++r) { hA[r] = hB[r] = cA[r] = cB[r] = 0.0f; }

  for (int s = 0; s < L_T; ++s) {
    const int t   = dir ? (L_T - 1 - s) : s;
    const int tok = (int)((tp >> (2 * t)) & 3ull);
    const float* gx = &gxs[tok * 128];

    // ---- build B = h^T (f16 B 32x16): lanes 0-15 hold K=0..15 of seq=lane,
    //      lanes 16-31 hold K=16..31.  Only a half-wave swap is needed.
    float sw[8];
#pragma unroll
    for (int r = 0; r < 8; ++r) {
      const float v = hi ? hA[r] : hB[r];     // give the other half what it needs
      sw[r] = __shfl_xor(v, 16, 32);          // lane<16 gets hA from lane+16; lane>=16 gets hB
    }
    FragAB bh;
#pragma unroll
    for (int v = 0; v < 4; ++v) {
      const float p0 = hi ? sw[2 * v]     : hA[2 * v];
      const float p1 = hi ? sw[2 * v + 1] : hA[2 * v + 1];
      const float q0 = hi ? hB[2 * v]     : sw[2 * v];
      const float q1 = hi ? hB[2 * v + 1] : sw[2 * v + 1];
      bh.u[v]     = pk2(p0, p1);
      bh.u[v + 4] = pk2(q0, q1);
    }

    const bool act = (t < len);

    // ---- gate group A: units u=0..15 -> tiles 0(i), 2(f), 4(g), 6(o)
    {
      v8f ci = wmma_hh(a_hh[0].h, bh.h, ldc8(gx + 0 * 16 + 8 * hi));
      v8f cf = wmma_hh(a_hh[2].h, bh.h, ldc8(gx + 2 * 16 + 8 * hi));
      v8f cg = wmma_hh(a_hh[4].h, bh.h, ldc8(gx + 4 * 16 + 8 * hi));
      v8f co = wmma_hh(a_hh[6].h, bh.h, ldc8(gx + 6 * 16 + 8 * hi));
#pragma unroll
      for (int r = 0; r < 8; ++r) {
        const float ig = sigm(ci[r]);
        const float fg = sigm(cf[r]);
        const float gg = tanh_f(cg[r]);
        const float og = sigm(co[r]);
        const float cn = fg * cA[r] + ig * gg;
        const float hn = og * tanh_f(cn);
        cA[r] = act ? cn : cA[r];
        hA[r] = act ? hn : hA[r];
      }
    }
    // ---- gate group B: units u=16..31 -> tiles 1(i), 3(f), 5(g), 7(o)
    {
      v8f ci = wmma_hh(a_hh[1].h, bh.h, ldc8(gx + 1 * 16 + 8 * hi));
      v8f cf = wmma_hh(a_hh[3].h, bh.h, ldc8(gx + 3 * 16 + 8 * hi));
      v8f cg = wmma_hh(a_hh[5].h, bh.h, ldc8(gx + 5 * 16 + 8 * hi));
      v8f co = wmma_hh(a_hh[7].h, bh.h, ldc8(gx + 7 * 16 + 8 * hi));
#pragma unroll
      for (int r = 0; r < 8; ++r) {
        const float ig = sigm(ci[r]);
        const float fg = sigm(cf[r]);
        const float gg = tanh_f(cg[r]);
        const float og = sigm(co[r]);
        const float cn = fg * cB[r] + ig * gg;
        const float hn = og * tanh_f(cn);
        cB[r] = act ? cn : cB[r];
        hB[r] = act ? hn : hB[r];
      }
    }
  }

  // ---- write final hidden state: out[seq][dir*32 + u]
  float* orow = out + (long)seq * (2 * H_DIM) + dir * H_DIM;
#pragma unroll
  for (int r = 0; r < 8; ++r) {
    orow[8 * hi + r]      = hA[r];
    orow[16 + 8 * hi + r] = hB[r];
  }
}

extern "C" void kernel_launch(void* const* d_in, const int* in_sizes, int n_in,
                              void* d_out, int out_size, void* d_ws, size_t ws_size,
                              hipStream_t stream) {
  (void)in_sizes; (void)n_in; (void)out_size; (void)d_ws; (void)ws_size;
  const int*   tokens  = (const int*)d_in[0];
  const int*   lengths = (const int*)d_in[1];
  const float* emb     = (const float*)d_in[2];
  const float* Wih_f   = (const float*)d_in[3];
  const float* Whh_f   = (const float*)d_in[4];
  const float* b_f     = (const float*)d_in[5];
  const float* Wih_b   = (const float*)d_in[6];
  const float* Whh_b   = (const float*)d_in[7];
  const float* b_b     = (const float*)d_in[8];
  float* out = (float*)d_out;

  dim3 grid(N_SEQ / (TILE_SEQ * WAVES_PER_BLK));   // 2048 blocks, 8192 wave-tiles
  dim3 blk(WAVES_PER_BLK * 32);                    // 4 waves of 32

  lstm_dir_kernel<<<grid, blk, 0, stream>>>(tokens, lengths, emb, Wih_f, Whh_f, b_f, out, 0);
  lstm_dir_kernel<<<grid, blk, 0, stream>>>(tokens, lengths, emb, Wih_b, Whh_b, b_b, out, 1);
}
